// QuantizedLinear_11579231830187
// MI455X (gfx1250) — matrix operand order repaired and checked by reading, not compiled, not run on hardware
//
#include <hip/hip_runtime.h>

// ---------------------------------------------------------------------------
// QuantizedLinear on MI455X (gfx1250):
//   y[b,s,o] = sum_k x[b,s,k] * W[o,k] + bias[o],  W = (q-8)*scale + U@V
//
// Stage 0: cvt_x_kernel      -> x  as bf16 [8192][4096] in d_ws (+32MB)
// Stage 1: dequant_w_kernel  -> W  as bf16 [4096][4096] in d_ws (offset 0)
// Stage 2: gemm_wmma_kernel  -> bf16 WMMA 16x16x32, f32 accumulate.
//          Block tile 256x128x64, 8 waves (4M x 2N), wave tile 64x64,
//          double-buffered dynamic LDS (96KB) filled by
//          global_load_async_to_lds_b128 (ASYNCcnt).
// ---------------------------------------------------------------------------

typedef __attribute__((ext_vector_type(16))) __bf16 v16bf;
typedef __attribute__((ext_vector_type(8)))  float  v8f;
typedef __attribute__((ext_vector_type(8)))  unsigned short us8;
typedef __attribute__((ext_vector_type(4)))  int    v4i;

union BFrag { us8 h[2]; v16bf v; };

constexpr int M_TOTAL = 8192;   // B * S
constexpr int N_TOTAL = 4096;   // OUT
constexpr int K_TOTAL = 4096;   // IN
constexpr int RANK    = 32;
constexpr int NGROUPS = 32;     // K / 128

constexpr int BM  = 256;
constexpr int BN  = 128;
constexpr int BKT = 64;         // K tile (two 32-wide WMMA k-steps)

constexpr size_t LDS_BYTES = (size_t)2 * (BM + BN) * BKT * 2;  // 96 KB

#if defined(__has_builtin)
#if __has_builtin(__builtin_amdgcn_global_load_async_to_lds_b128) && \
    __has_builtin(__builtin_amdgcn_s_wait_asynccnt)
#define HAVE_ASYNC_LDS 1
#endif
#endif

__device__ __forceinline__ unsigned short f2bf(float f) {
  unsigned int u = __float_as_uint(f);
  u += 0x7FFFu + ((u >> 16) & 1u);
  return (unsigned short)(u >> 16);
}

// ---------------------------------------------------------------------------
// Stage 0: x f32 -> bf16. One block per (b,s) row, 16 elems/thread.
// ---------------------------------------------------------------------------
__global__ __launch_bounds__(256)
void cvt_x_kernel(const float* __restrict__ x, unsigned short* __restrict__ Xb) {
  const long base = (long)blockIdx.x * K_TOTAL + threadIdx.x * 16;
  const float4* xp = reinterpret_cast<const float4*>(x + base);
  unsigned short res[16];
#pragma unroll
  for (int j = 0; j < 4; ++j) {
    const float4 v = xp[j];
    res[4 * j + 0] = f2bf(v.x);
    res[4 * j + 1] = f2bf(v.y);
    res[4 * j + 2] = f2bf(v.z);
    res[4 * j + 3] = f2bf(v.w);
  }
  us8* dst = reinterpret_cast<us8*>(Xb + base);
  dst[0] = *reinterpret_cast<us8*>(&res[0]);
  dst[1] = *reinterpret_cast<us8*>(&res[8]);
}

// ---------------------------------------------------------------------------
// Stage 1: W_bf16[n][k] = (q[n][k]-8)*scales[n][k/128] + sum_r U[n][r]*V[r][k]
// ---------------------------------------------------------------------------
__global__ __launch_bounds__(256)
void dequant_w_kernel(const int* __restrict__ q,
                      const float* __restrict__ scales,
                      const float* __restrict__ U,
                      const float* __restrict__ V,
                      unsigned short* __restrict__ Wb) {
  const int n  = blockIdx.x;
  const int k0 = threadIdx.x * 16;
  const float scale = scales[n * NGROUPS + (k0 >> 7)];

  float acc[16];
#pragma unroll
  for (int i = 0; i < 16; ++i) acc[i] = 0.0f;

  const float* __restrict__ Urow = U + n * RANK;
#pragma unroll 8
  for (int r = 0; r < RANK; ++r) {
    const float u = Urow[r];
    const float4* vp = reinterpret_cast<const float4*>(V + (long)r * K_TOTAL + k0);
#pragma unroll
    for (int j = 0; j < 4; ++j) {
      const float4 vv = vp[j];
      acc[4 * j + 0] = fmaf(u, vv.x, acc[4 * j + 0]);
      acc[4 * j + 1] = fmaf(u, vv.y, acc[4 * j + 1]);
      acc[4 * j + 2] = fmaf(u, vv.z, acc[4 * j + 2]);
      acc[4 * j + 3] = fmaf(u, vv.w, acc[4 * j + 3]);
    }
  }

  const int4* qp = reinterpret_cast<const int4*>(q + (long)n * K_TOTAL + k0);
  unsigned short res[16];
#pragma unroll
  for (int j = 0; j < 4; ++j) {
    const int4 qq = qp[j];
    res[4 * j + 0] = f2bf(fmaf((float)(qq.x - 8), scale, acc[4 * j + 0]));
    res[4 * j + 1] = f2bf(fmaf((float)(qq.y - 8), scale, acc[4 * j + 1]));
    res[4 * j + 2] = f2bf(fmaf((float)(qq.z - 8), scale, acc[4 * j + 2]));
    res[4 * j + 3] = f2bf(fmaf((float)(qq.w - 8), scale, acc[4 * j + 3]));
  }

  us8* dst = reinterpret_cast<us8*>(Wb + (long)n * K_TOTAL + k0);
  dst[0] = *reinterpret_cast<us8*>(&res[0]);
  dst[1] = *reinterpret_cast<us8*>(&res[8]);
}

// ---------------------------------------------------------------------------
// Stage 2: tiled GEMM, wave tile 64x64 = 4x4 WMMA tiles.
// ---------------------------------------------------------------------------
__global__ __launch_bounds__(256)
void gemm_wmma_kernel(const unsigned short* __restrict__ Xb,
                      const unsigned short* __restrict__ Wb,
                      const float* __restrict__ bias,
                      float* __restrict__ out) {
  extern __shared__ unsigned short smem[];
  // As: [2][BM][BKT] (64 KB), Bs: [2][BN][BKT] (32 KB)
  unsigned short(*As)[BM][BKT] = reinterpret_cast<unsigned short(*)[BM][BKT]>(smem);
  unsigned short(*Bs)[BN][BKT] =
      reinterpret_cast<unsigned short(*)[BN][BKT]>(smem + 2 * BM * BKT);

  const int t        = threadIdx.x;
  const int lane     = t & 31;
  const int wave     = t >> 5;
  const int waveM    = wave >> 1;             // 4 waves along M (64 rows each)
  const int waveN    = wave & 1;              // 2 waves along N (64 cols each)
  const int laneHalf = lane & 15;
  const int laneHi   = lane >> 4;

  const int mBase = blockIdx.y * BM;
  const int nBase = blockIdx.x * BN;

  // staging: A = one full 64-elem row per thread (8 x b128);
  //          B = half row per thread (4 x b128)
  const int aRow = t;                         // 0..255
  const int bRow = t >> 1;                    // 0..127
  const int bK   = (t & 1) * 32;

  const unsigned short* __restrict__ xrow = Xb + (long)(mBase + aRow) * K_TOTAL;
  const unsigned short* __restrict__ wrow = Wb + (long)(nBase + bRow) * K_TOTAL + bK;

  v8f acc[4][4];
#pragma unroll
  for (int mi = 0; mi < 4; ++mi)
#pragma unroll
    for (int ni = 0; ni < 4; ++ni)
      acc[mi][ni] = (v8f){0.f, 0.f, 0.f, 0.f, 0.f, 0.f, 0.f, 0.f};

  auto stage = [&](int buf, int kk) {
#ifdef HAVE_ASYNC_LDS
#pragma unroll
    for (int j = 0; j < 8; ++j)
      __builtin_amdgcn_global_load_async_to_lds_b128(
          (v4i*)(xrow + kk + j * 8), (v4i*)&As[buf][aRow][j * 8], 0, 0);
#pragma unroll
    for (int j = 0; j < 4; ++j)
      __builtin_amdgcn_global_load_async_to_lds_b128(
          (v4i*)(wrow + kk + j * 8), (v4i*)&Bs[buf][bRow][bK + j * 8], 0, 0);
#else
    const us8* xs = reinterpret_cast<const us8*>(xrow + kk);
    const us8* ws = reinterpret_cast<const us8*>(wrow + kk);
#pragma unroll
    for (int j = 0; j < 8; ++j)
      *reinterpret_cast<us8*>(&As[buf][aRow][j * 8]) = xs[j];
#pragma unroll
    for (int j = 0; j < 4; ++j)
      *reinterpret_cast<us8*>(&Bs[buf][bRow][bK + j * 8]) = ws[j];
#endif
  };

  auto compute = [&](int buf) {
#pragma unroll
    for (int s = 0; s < 2; ++s) {            // two 32-wide k-steps
      const int kb = s * 32;
      BFrag a[4], b[4];
      // A 16x32 bf16: lane<16 -> K 0-7 & 16-23; lane>=16 -> K 8-15 & 24-31
#pragma unroll
      for (int mi = 0; mi < 4; ++mi) {
        const int row = waveM * 64 + mi * 16 + laneHalf;
        const int kO  = kb + laneHi * 8;
        a[mi].h[0] = *reinterpret_cast<const us8*>(&As[buf][row][kO + 0]);
        a[mi].h[1] = *reinterpret_cast<const us8*>(&As[buf][row][kO + 16]);
      }
      // B 32x16 bf16: lane<16 -> K 0-7 & 16-23; lane>=16 -> K 8-15 & 24-31
#pragma unroll
      for (int ni = 0; ni < 4; ++ni) {
        const int col = waveN * 64 + ni * 16 + laneHalf;
        const int kB  = kb + laneHi * 8;
        b[ni].h[0] = *reinterpret_cast<const us8*>(&Bs[buf][col][kB + 0]);
        b[ni].h[1] = *reinterpret_cast<const us8*>(&Bs[buf][col][kB + 16]);
      }
#pragma unroll
      for (int mi = 0; mi < 4; ++mi)
#pragma unroll
        for (int ni = 0; ni < 4; ++ni)
          acc[mi][ni] = __builtin_amdgcn_wmma_f32_16x16x32_bf16(
              false, a[mi].v, false, b[ni].v,
              (short)0, acc[mi][ni], false, false);
    }
  };

  // -------- main loop: ping-pong buffers -------------------------------
  stage(0, 0);
#ifdef HAVE_ASYNC_LDS
  __builtin_amdgcn_s_wait_asynccnt(0);
#endif
  __syncthreads();

  int cur = 0;
  for (int kk = 0; kk < K_TOTAL; kk += BKT) {
    const int nxt = cur ^ 1;
    if (kk + BKT < K_TOTAL) stage(nxt, kk + BKT);   // fill next while computing
    compute(cur);
#ifdef HAVE_ASYNC_LDS
    __builtin_amdgcn_s_wait_asynccnt(0);            // my async fills done
#endif
    __syncthreads();                                // everyone's fills done
    cur = nxt;
  }

  // -------- epilogue: C/D layout VGPR r: lane<16 -> M=r, lane>=16 -> M=8+r
#pragma unroll
  for (int mi = 0; mi < 4; ++mi) {
#pragma unroll
    for (int ni = 0; ni < 4; ++ni) {
      const int ncol  = nBase + waveN * 64 + ni * 16 + laneHalf;
      const int mrow0 = mBase + waveM * 64 + mi * 16 + laneHi * 8;
      const float bv  = bias[ncol];
      float* op = out + (long)mrow0 * N_TOTAL + ncol;
#pragma unroll
      for (int r = 0; r < 8; ++r)
        op[(long)r * N_TOTAL] = acc[mi][ni][r] + bv;
    }
  }
}

// ---------------------------------------------------------------------------
extern "C" void kernel_launch(void* const* d_in, const int* in_sizes, int n_in,
                              void* d_out, int out_size, void* d_ws, size_t ws_size,
                              hipStream_t stream) {
  const float* x      = (const float*)d_in[0];
  const int*   q      = (const int*)d_in[1];
  const float* scales = (const float*)d_in[2];
  const float* U      = (const float*)d_in[3];
  const float* V      = (const float*)d_in[4];
  const float* bias   = (const float*)d_in[5];
  float*       out    = (float*)d_out;

  unsigned short* Wb = (unsigned short*)d_ws;                       // 32 MB
  unsigned short* Xb = (unsigned short*)((char*)d_ws + (size_t)N_TOTAL * K_TOTAL * 2);

  dequant_w_kernel<<<dim3(N_TOTAL), dim3(256), 0, stream>>>(q, scales, U, V, Wb);
  cvt_x_kernel<<<dim3(M_TOTAL), dim3(256), 0, stream>>>(x, Xb);
  gemm_wmma_kernel<<<dim3(N_TOTAL / BN, M_TOTAL / BM), dim3(256),
                     (unsigned int)LDS_BYTES, stream>>>(Xb, Wb, bias, out);
}
